// Model_20598663152059
// MI455X (gfx1250) — matrix operand order repaired and checked by reading, not compiled, not run on hardware
//
#include <hip/hip_runtime.h>

// Problem constants (from reference)
#define LAYERS 9
#define CIN 3
#define CCH 256     // C
#define KPART 3     // K
#define VJ 25       // V
#define GW 9        // G (temporal window)
#define TT_ 128     // T
#define NCLS 60     // NC
#define KC (KPART*CCH)        // 768
#define NCOL (TT_*VJ)         // 3200
#define NTILES (NCOL/16)      // 200
#define ZROWS ((TT_+GW-1)*VJ) // 3400 (8 pad frames + 128 frames)

typedef __attribute__((ext_vector_type(16))) __bf16 v16bf;
typedef __attribute__((ext_vector_type(8)))  float  v8f;
typedef __attribute__((ext_vector_type(4)))  float  vf4;

// ---------------- WMMA fragment loaders (wave32, gfx1250 layouts) ----------------
// A: 16x32 bf16 tile, row-major, lda elements per row.
// lanes 0-15: M=lane, VGPR0-3 = K kk+0..7, VGPR4-7 = K kk+16..23
// lanes16-31: M=lane-16, VGPR0-3 = K kk+8..15, VGPR4-7 = K kk+24..31
__device__ __forceinline__ v16bf frag_a(const __bf16* Ab, int lda, int kk, int lane) {
    int m  = lane & 15;
    int hi = lane >> 4;
    union { v16bf v; vf4 f[2]; } u;
    const __bf16* p = Ab + (size_t)m * lda + kk + hi * 8;
    u.f[0] = *(const vf4*)p;          // 8 bf16 = 16B
    u.f[1] = *(const vf4*)(p + 16);   // next K-group
    return u.v;
}
// B: 32x16 bf16 tile read from B^T (row n holds K contiguous), ldb elements per row.
// lanes 0-15: N=lane, K kk+0..7,16..23 ; lanes16-31: N=lane-16, K kk+8..15,24..31
__device__ __forceinline__ v16bf frag_b(const __bf16* Bt, size_t row0, int ldb, int kk, int lane) {
    int n  = lane & 15;
    int kb = (lane >> 4) * 8;
    union { v16bf v; vf4 f[2]; } u;
    const __bf16* p = Bt + (row0 + n) * (size_t)ldb + kk + kb;
    u.f[0] = *(const vf4*)p;
    u.f[1] = *(const vf4*)(p + 16);
    return u.v;
}
// D tile store (C/D layout: M = i + 8*(lane>=16), N = lane&15), output transposed [n][m]
__device__ __forceinline__ void store_d(float* O, int ldo, int ntile, int mtile, int lane,
                                        v8f acc) {
    int n  = ntile * 16 + (lane & 15);
    int mb = mtile * 16 + (lane >> 4) * 8;
    float* o = O + (size_t)n * ldo + mb;
    vf4 lo = {acc[0], acc[1], acc[2], acc[3]};
    vf4 hi = {acc[4], acc[5], acc[6], acc[7]};
    *(vf4*)o = lo;
    *(vf4*)(o + 4) = hi;
}
__device__ __forceinline__ void store_d_bias(float* O, int ldo, int ntile, int mtile, int lane,
                                             v8f acc, const float* bias) {
    int n  = ntile * 16 + (lane & 15);
    int mb = mtile * 16 + (lane >> 4) * 8;
    const float* bp = bias + mb;
    float* o = O + (size_t)n * ldo + mb;
    vf4 lo = {acc[0] + bp[0], acc[1] + bp[1], acc[2] + bp[2], acc[3] + bp[3]};
    vf4 hi = {acc[4] + bp[4], acc[5] + bp[5], acc[6] + bp[6], acc[7] + bp[7]};
    *(vf4*)o = lo;
    *(vf4*)(o + 4) = hi;
}

#define WMMA_BF16(A, B, CACC) \
    __builtin_amdgcn_wmma_f32_16x16x32_bf16(false, (A), false, (B), (short)0, (CACC), false, false)

// ---------------- weight conversion kernels (fp32 -> bf16, once per launch) ------
__global__ void k_cvt(const float* __restrict__ in, __bf16* __restrict__ out, int n) {
    int i = blockIdx.x * blockDim.x + threadIdx.x;
    if (i < n) out[i] = (__bf16)in[i];
}
// Wt (L,C,C,G) -> bf16 [L][G][O][C]
__global__ void k_cvt_wt(const float* __restrict__ Wt, __bf16* __restrict__ out) {
    int i = blockIdx.x * blockDim.x + threadIdx.x;
    const int total = LAYERS * GW * CCH * CCH;
    if (i >= total) return;
    int c = i & (CCH - 1);
    int o = (i / CCH) & (CCH - 1);
    int g = (i / (CCH * CCH)) % GW;
    int l = i / (CCH * CCH * GW);
    out[i] = (__bf16)Wt[(((size_t)l * CCH + o) * CCH + c) * GW + g];
}

// ---------------- input LN + fcn_in ----------------------------------------------
__global__ void k_input(const float* __restrict__ x, const float* __restrict__ lnw,
                        const float* __restrict__ lnb, const float* __restrict__ Win,
                        const float* __restrict__ bin,
                        float* __restrict__ Hf, __bf16* __restrict__ Hb) {
    __shared__ float xr[CIN * VJ];
    __shared__ float xn[CIN * VJ];
    __shared__ float st[2];
    int t = blockIdx.x, tid = threadIdx.x;
    if (tid < CIN * VJ) {
        int ci = tid / VJ, v = tid % VJ;
        xr[tid] = x[((size_t)ci * TT_ + t) * VJ + v];
    }
    __syncthreads();
    if (tid == 0) {
        float s = 0.f, s2 = 0.f;
        for (int i = 0; i < CIN * VJ; ++i) { s += xr[i]; s2 += xr[i] * xr[i]; }
        float m = s / (CIN * VJ);
        st[0] = m; st[1] = rsqrtf(s2 / (CIN * VJ) - m * m + 1e-5f);
    }
    __syncthreads();
    if (tid < CIN * VJ) xn[tid] = (xr[tid] - st[0]) * st[1] * lnw[tid] + lnb[tid];
    __syncthreads();
    int c = tid;
    for (int v = 0; v < VJ; ++v) {
        float acc = bin[c];
        for (int ci = 0; ci < CIN; ++ci) acc += Win[c * CIN + ci] * xn[ci * VJ + v];
        size_t idx = (size_t)(t * VJ + v) * CCH + c;
        Hf[idx] = acc;
        Hb[idx] = (__bf16)acc;
    }
}

// ---------------- GEMM1: Y^T = (Wg (768x256) * H (256x3200))^T, 2x2 reg blocking --
__global__ __launch_bounds__(128) void k_gemm_graph(const __bf16* __restrict__ Wgbf,
                                                    const __bf16* __restrict__ Hb,
                                                    float* __restrict__ YT, int l) {
    int lane = threadIdx.x & 31;
    int wave = threadIdx.x >> 5;
    int mb = blockIdx.y * 4 + wave;      // 0..23 (pairs of M-tiles)
    int nb = blockIdx.x;                 // 0..99 (pairs of N-tiles)
    const __bf16* A0 = Wgbf + (size_t)l * KC * CCH + (size_t)mb * 32 * CCH;
    size_t nrow0 = (size_t)nb * 32;
    v8f a00 = {}, a01 = {}, a10 = {}, a11 = {};
    for (int kk = 0; kk < CCH; kk += 32) {
        v16bf fa0 = frag_a(A0, CCH, kk, lane);
        v16bf fa1 = frag_a(A0 + (size_t)16 * CCH, CCH, kk, lane);
        v16bf fb0 = frag_b(Hb, nrow0, CCH, kk, lane);
        v16bf fb1 = frag_b(Hb, nrow0 + 16, CCH, kk, lane);
        a00 = WMMA_BF16(fa0, fb0, a00);
        a01 = WMMA_BF16(fa0, fb1, a01);
        a10 = WMMA_BF16(fa1, fb0, a10);
        a11 = WMMA_BF16(fa1, fb1, a11);
    }
    store_d(YT, KC, nb * 2,     mb * 2,     lane, a00);
    store_d(YT, KC, nb * 2 + 1, mb * 2,     lane, a01);
    store_d(YT, KC, nb * 2,     mb * 2 + 1, lane, a10);
    store_d(YT, KC, nb * 2 + 1, mb * 2 + 1, lane, a11);
}

// ---------------- graph einsum + window LN + ReLU -> ZnT (bf16, padded) ----------
__global__ void k_graph(const float* __restrict__ YT, const float* __restrict__ Aadj,
                        const float* __restrict__ imp, const float* __restrict__ bg,
                        const float* __restrict__ ln1w, const float* __restrict__ ln1b,
                        __bf16* __restrict__ ZnT, int l) {
    __shared__ float sAl[KPART * VJ * VJ];
    __shared__ float red[CCH], red2[CCH];
    int t = blockIdx.x, c = threadIdx.x;
    for (int i = c; i < KPART * VJ * VJ; i += blockDim.x)
        sAl[i] = Aadj[i] * imp[(size_t)l * KPART * VJ * VJ + i];
    __syncthreads();
    float z[VJ];
    for (int w = 0; w < VJ; ++w) z[w] = 0.f;
    for (int k = 0; k < KPART; ++k) {
        float bgv = bg[(size_t)l * KC + k * CCH + c];
        for (int v = 0; v < VJ; ++v) {
            float yv = YT[(size_t)(t * VJ + v) * KC + k * CCH + c] + bgv;
            const float* ar = &sAl[(k * VJ + v) * VJ];
            for (int w = 0; w < VJ; ++w) z[w] += yv * ar[w];
        }
    }
    float s = 0.f, s2 = 0.f;
    for (int w = 0; w < VJ; ++w) { s += z[w]; s2 += z[w] * z[w]; }
    red[c] = s; red2[c] = s2;
    __syncthreads();
    for (int off = CCH / 2; off > 0; off >>= 1) {
        if (c < off) { red[c] += red[c + off]; red2[c] += red2[c + off]; }
        __syncthreads();
    }
    float mean = red[0] / (CCH * VJ);
    float rs = rsqrtf(red2[0] / (CCH * VJ) - mean * mean + 1e-5f);
    for (int w = 0; w < VJ; ++w) {
        float a = (z[w] - mean) * rs * ln1w[(size_t)l * CCH * VJ + c * VJ + w] +
                  ln1b[(size_t)l * CCH * VJ + c * VJ + w];
        a = fmaxf(a, 0.f);
        ZnT[(size_t)((t + GW - 1) * VJ + w) * CCH + c] = (__bf16)a;
    }
}

// history pad frames: LN(0) = ln1_b -> relu
__global__ void k_pad(const float* __restrict__ ln1b, __bf16* __restrict__ ZnT, int l) {
    int i = blockIdx.x * blockDim.x + threadIdx.x;
    const int total = (GW - 1) * VJ * CCH;
    if (i >= total) return;
    int c = i & (CCH - 1);
    int row = i / CCH;
    int w = row % VJ;
    float v = fmaxf(ln1b[(size_t)l * CCH * VJ + c * VJ + w], 0.f);
    ZnT[(size_t)row * CCH + c] = (__bf16)v;
}

// ---------------- GEMM2: 9-tap temporal conv fused, 2x2 reg blocking --------------
__global__ __launch_bounds__(128) void k_gemm_tcn(const __bf16* __restrict__ Wtbf,
                                                  const __bf16* __restrict__ ZnT,
                                                  const float* __restrict__ bt,
                                                  float* __restrict__ TTo, int l) {
    int lane = threadIdx.x & 31;
    int wave = threadIdx.x >> 5;
    int mb = blockIdx.y * 4 + wave;      // 0..7 (pairs of M-tiles)
    int nb = blockIdx.x;                 // 0..99 (pairs of N-tiles)
    v8f a00 = {}, a01 = {}, a10 = {}, a11 = {};
    for (int g = 0; g < GW; ++g) {
        const __bf16* A0 = Wtbf + ((size_t)(l * GW + g) * CCH) * CCH + (size_t)mb * 32 * CCH;
        size_t row0 = (size_t)nb * 32 + (size_t)(GW - 1 - g) * VJ;  // frame (t-g)
        for (int kk = 0; kk < CCH; kk += 32) {
            v16bf fa0 = frag_a(A0, CCH, kk, lane);
            v16bf fa1 = frag_a(A0 + (size_t)16 * CCH, CCH, kk, lane);
            v16bf fb0 = frag_b(ZnT, row0, CCH, kk, lane);
            v16bf fb1 = frag_b(ZnT, row0 + 16, CCH, kk, lane);
            a00 = WMMA_BF16(fa0, fb0, a00);
            a01 = WMMA_BF16(fa0, fb1, a01);
            a10 = WMMA_BF16(fa1, fb0, a10);
            a11 = WMMA_BF16(fa1, fb1, a11);
        }
    }
    const float* bias = bt + (size_t)l * CCH;
    store_d_bias(TTo, CCH, nb * 2,     mb * 2,     lane, a00, bias);
    store_d_bias(TTo, CCH, nb * 2 + 1, mb * 2,     lane, a01, bias);
    store_d_bias(TTo, CCH, nb * 2,     mb * 2 + 1, lane, a10, bias);
    store_d_bias(TTo, CCH, nb * 2 + 1, mb * 2 + 1, lane, a11, bias);
}

// ---------------- LN2 + center residual + ReLU -> next H --------------------------
__global__ void k_post(const float* __restrict__ TTi, const float* __restrict__ ln2w,
                       const float* __restrict__ ln2b, const float* __restrict__ HfIn,
                       float* __restrict__ HfOut, __bf16* __restrict__ HbOut, int l) {
    __shared__ float red[CCH], red2[CCH];
    int t = blockIdx.x, c = threadIdx.x;
    float xv[VJ];
    float s = 0.f, s2 = 0.f;
    for (int v = 0; v < VJ; ++v) {
        float val = TTi[(size_t)(t * VJ + v) * CCH + c];
        xv[v] = val; s += val; s2 += val * val;
    }
    red[c] = s; red2[c] = s2;
    __syncthreads();
    for (int off = CCH / 2; off > 0; off >>= 1) {
        if (c < off) { red[c] += red[c + off]; red2[c] += red2[c + off]; }
        __syncthreads();
    }
    float mean = red[0] / (CCH * VJ);
    float rs = rsqrtf(red2[0] / (CCH * VJ) - mean * mean + 1e-5f);
    for (int v = 0; v < VJ; ++v) {
        float t2 = (xv[v] - mean) * rs * ln2w[(size_t)l * CCH * VJ + c * VJ + v] +
                   ln2b[(size_t)l * CCH * VJ + c * VJ + v];
        float r = (t >= GW / 2) ? HfIn[(size_t)((t - GW / 2) * VJ + v) * CCH + c] : 0.f;
        float h = fmaxf(t2 + r, 0.f);
        size_t idx = (size_t)(t * VJ + v) * CCH + c;
        HfOut[idx] = h;
        HbOut[idx] = (__bf16)h;
    }
}

// ---------------- head: pool over V + classifier ---------------------------------
__global__ void k_head(const float* __restrict__ Hf, const float* __restrict__ Wout,
                       const float* __restrict__ bout, float* __restrict__ out) {
    __shared__ float pooled[CCH];
    int t = blockIdx.x, c = threadIdx.x;
    float s = 0.f;
    for (int v = 0; v < VJ; ++v) s += Hf[(size_t)(t * VJ + v) * CCH + c];
    pooled[c] = s * (1.0f / VJ);
    __syncthreads();
    if (c < NCLS) {
        float acc = bout[c];
        for (int k = 0; k < CCH; ++k) acc += Wout[c * CCH + k] * pooled[k];
        out[t * NCLS + c] = acc;
    }
}

// ---------------- host orchestration ---------------------------------------------
extern "C" void kernel_launch(void* const* d_in, const int* in_sizes, int n_in,
                              void* d_out, int out_size, void* d_ws, size_t ws_size,
                              hipStream_t stream) {
    const float* x      = (const float*)d_in[0];
    const float* Aadj   = (const float*)d_in[1];
    const float* lninw  = (const float*)d_in[2];
    const float* lninb  = (const float*)d_in[3];
    const float* W_in   = (const float*)d_in[4];
    const float* b_in   = (const float*)d_in[5];
    const float* Wg     = (const float*)d_in[6];
    const float* bg     = (const float*)d_in[7];
    const float* ln1w   = (const float*)d_in[8];
    const float* ln1b   = (const float*)d_in[9];
    const float* Wt     = (const float*)d_in[10];
    const float* bt     = (const float*)d_in[11];
    const float* ln2w   = (const float*)d_in[12];
    const float* ln2b   = (const float*)d_in[13];
    const float* imp    = (const float*)d_in[14];
    const float* W_out  = (const float*)d_in[15];
    const float* b_out  = (const float*)d_in[16];
    float* out = (float*)d_out;

    char* ws = (char*)d_ws;
    size_t off = 0;
    auto alloc = [&](size_t bytes) -> void* {
        void* p = ws + off;
        off = (off + bytes + 255) & ~(size_t)255;
        return p;
    };
    __bf16* Wgbf = (__bf16*)alloc((size_t)LAYERS * KC * CCH * 2);
    __bf16* Wtbf = (__bf16*)alloc((size_t)LAYERS * GW * CCH * CCH * 2);
    float*  HfA  = (float*)alloc((size_t)NCOL * CCH * 4);
    float*  HfB  = (float*)alloc((size_t)NCOL * CCH * 4);
    __bf16* HbA  = (__bf16*)alloc((size_t)NCOL * CCH * 2);
    __bf16* HbB  = (__bf16*)alloc((size_t)NCOL * CCH * 2);
    float*  YT   = (float*)alloc((size_t)NCOL * KC * 4);
    __bf16* ZnT  = (__bf16*)alloc((size_t)ZROWS * CCH * 2);
    float*  TTb  = (float*)alloc((size_t)NCOL * CCH * 4);

    // weight conversion (deterministic each call)
    {
        int n = LAYERS * KC * CCH;
        k_cvt<<<dim3((n + 255) / 256), dim3(256), 0, stream>>>(Wg, Wgbf, n);
        int m = LAYERS * GW * CCH * CCH;
        k_cvt_wt<<<dim3((m + 255) / 256), dim3(256), 0, stream>>>(Wt, Wtbf);
    }

    float*  HfCur = HfA; float*  HfNxt = HfB;
    __bf16* HbCur = HbA; __bf16* HbNxt = HbB;

    k_input<<<dim3(TT_), dim3(CCH), 0, stream>>>(x, lninw, lninb, W_in, b_in, HfCur, HbCur);

    for (int l = 0; l < LAYERS; ++l) {
        // GEMM1: 48 M-tiles -> 24 pairs -> grid.y=6 (4 waves/block); 200 N-tiles -> 100 pairs
        k_gemm_graph<<<dim3(NTILES / 2, 6), dim3(128), 0, stream>>>(Wgbf, HbCur, YT, l);
        k_graph<<<dim3(TT_), dim3(CCH), 0, stream>>>(YT, Aadj, imp, bg, ln1w, ln1b, ZnT, l);
        {
            int n = (GW - 1) * VJ * CCH;
            k_pad<<<dim3((n + 255) / 256), dim3(256), 0, stream>>>(ln1b, ZnT, l);
        }
        // GEMM2: 16 M-tiles -> 8 pairs -> grid.y=2
        k_gemm_tcn<<<dim3(NTILES / 2, 2), dim3(128), 0, stream>>>(Wtbf, ZnT, bt, TTb, l);
        k_post<<<dim3(TT_), dim3(CCH), 0, stream>>>(TTb, ln2w, ln2b, HfCur, HfNxt, HbNxt, l);
        { float* tf = HfCur; HfCur = HfNxt; HfNxt = tf; }
        { __bf16* tb = HbCur; HbCur = HbNxt; HbNxt = tb; }
    }

    k_head<<<dim3(TT_), dim3(CCH), 0, stream>>>(HfCur, W_out, b_out, out);
}
